// FeatMatchNN_83502754169011
// MI455X (gfx1250) — matrix operand order repaired and checked by reading, not compiled, not run on hardware
//
#include <hip/hip_runtime.h>
#include <hip/hip_bf16.h>

typedef __attribute__((ext_vector_type(16))) _Float16 v16h;
typedef __attribute__((ext_vector_type(8)))  float    v8f;
typedef __attribute__((ext_vector_type(4)))  unsigned u32x4;
typedef __attribute__((ext_vector_type(8)))  int      i32x8;
typedef __attribute__((ext_vector_type(4)))  int      i32x4;

#define BCONST 2
#define LCONST 32
#define NEG_SLOPE 0.01f
#define EPSN 1e-8f

// ---------------------------------------------------------------------------
// Encoder: direct 5x5 conv (pad 2) + leaky ReLU  (bandwidth-trivial, VALU)
// ---------------------------------------------------------------------------
__global__ void conv5_leaky(const float* __restrict__ in, const float* __restrict__ w,
                            const float* __restrict__ bias, float* __restrict__ out,
                            int Bn, int Cin, int Cout, int Hh, int Ww)
{
    int idx = blockIdx.x * blockDim.x + threadIdx.x;
    int total = Bn * Cout * Hh * Ww;
    if (idx >= total) return;
    int x = idx % Ww; int t = idx / Ww;
    int y = t % Hh;   t /= Hh;
    int co = t % Cout; int b = t / Cout;
    float acc = bias[co];
    for (int ci = 0; ci < Cin; ++ci) {
        const float* ip = in + ((b * Cin + ci) * Hh) * Ww;
        const float* wp = w + ((co * Cin + ci) * 25);
        for (int ky = 0; ky < 5; ++ky) {
            int yy = y + ky - 2;
            if (yy < 0 || yy >= Hh) continue;
            for (int kx = 0; kx < 5; ++kx) {
                int xx = x + kx - 2;
                if (xx < 0 || xx >= Ww) continue;
                acc += ip[yy * Ww + xx] * wp[ky * 5 + kx];
            }
        }
    }
    out[idx] = acc >= 0.f ? acc : NEG_SLOPE * acc;
}

__global__ void maxpool2(const float* __restrict__ in, float* __restrict__ out,
                         int Bn, int C, int Ho, int Wo)
{
    int idx = blockIdx.x * blockDim.x + threadIdx.x;
    int total = Bn * C * Ho * Wo;
    if (idx >= total) return;
    int x = idx % Wo; int t = idx / Wo;
    int y = t % Ho;   t /= Ho;
    int c = t % C;    int b = t / C;
    int Wi = Wo * 2;
    const float* p = in + ((size_t)(b * C + c) * Ho * 2 + y * 2) * Wi + x * 2;
    out[idx] = fmaxf(fmaxf(p[0], p[1]), fmaxf(p[Wi], p[Wi + 1]));
}

// ---------------------------------------------------------------------------
// Attention level: cosine sim of every h2 pixel vs gathered h1 queries.
// P x L GEMM, K = C (compile-time 4/8/16) -> one zero-padded WMMA chunk.
// ---------------------------------------------------------------------------
template <int C>
__global__ void attn_wmma(const float* __restrict__ h1p, const float* __restrict__ h2p,
                          const int* __restrict__ pts, float* __restrict__ attnOut,
                          int Hs, int Ws, int shift)
{
    int lane = threadIdx.x & 31;
    int wv   = threadIdx.x >> 5;
    int P    = Hs * Ws;
    int pixBase = (blockIdx.x * 8 + wv) * 16;
    int b     = blockIdx.y >> 1;
    int lBase = (blockIdx.y & 1) * 16;

    // A fragment: h2 pixels, M = pixel, K = channel (16x32 f16 layout)
    int m    = lane & 15;
    int ksel = lane >> 4;
    int pixA = pixBase + m;
    v16h a;
#pragma unroll
    for (int t = 0; t < 16; ++t) {
        int kl = (t < 8) ? (ksel * 8 + t) : (16 + ksel * 8 + (t - 8));
        float av = (kl < C) ? h2p[(b * C + kl) * P + pixA] : 0.f;
        a[t] = (_Float16)av;
    }

    // B fragment: query vectors, K = channel, N = point (32x16 f16)
    int col  = lane & 15;
    int kOff = (lane >> 4) * 16;
    int l    = lBase + col;
    int py = pts[(b * LCONST + l) * 2 + 0] >> shift;
    int px = pts[(b * LCONST + l) * 2 + 1] >> shift;
    int qidx = py * Ws + px;
    v16h qb;
#pragma unroll
    for (int t = 0; t < 16; ++t) {
        int c = (t < 8) ? ((lane >> 4) * 8 + t) : (16 + (lane >> 4) * 8 + (t - 8));
        float qv = (c < C) ? h1p[(b * C + c) * P + qidx] : 0.f;
        qb[t] = (_Float16)qv;
    }

    v8f acc = {};
    acc = __builtin_amdgcn_wmma_f32_16x16x32_f16(false, a, false, qb,
                                                 (short)0, acc, false, false);

    float qn = 0.f;
#pragma unroll
    for (int c = 0; c < C; ++c) {
        float qv = h1p[(b * C + c) * P + qidx];
        qn += qv * qv;
    }
    qn = fmaxf(sqrtf(qn), EPSN);

    int hi = lane >> 4;
#pragma unroll
    for (int r = 0; r < 8; ++r) {
        int pix = pixBase + r + 8 * hi;   // D: M = r + 8*(lane>=16), N = lane&15
        float pn = 0.f;
#pragma unroll
        for (int c = 0; c < C; ++c) {
            float hv = h2p[(b * C + c) * P + pix];
            pn += hv * hv;
        }
        pn = fmaxf(sqrtf(pn), EPSN);
        attnOut[(size_t)(b * LCONST + l) * P + pix] = acc[r] / (pn * qn);
    }
}

// im2col patch offset for flat-k (constant-folds after loop unrolling)
__device__ __forceinline__ int patchOff(int k)
{
    int cin = k / 9;
    int r   = k - cin * 9;
    int kh  = r / 3;
    int kw  = r - kh * 3;
    return cin * (10 * 18) + kh * 18 + kw;
}

// ---------------------------------------------------------------------------
// Decoder conv as implicit GEMM with WMMA, fully LDS-staged.
// Block = 8x16 pixel tile (one row of 16 px per wave). The (CPAD x 10 x 18)
// f16 input halo patch is staged once (concat + 2x nearest upsample + zero
// borders baked in); weights staged as 16 x KPAD f16. The unrolled K loop is
// pure ds_load + v_wmma; per-element patch offsets are compile-time constants
// selected on the lane's K-half (one v_cndmask each), EXEC all ones.
// ---------------------------------------------------------------------------
template <int Cp, int Cf, int Cout>
__global__ void dec_conv_wmma(const float* __restrict__ prev,
                              const float* __restrict__ attn,
                              const float* __restrict__ feat,
                              const float* __restrict__ w,
                              const float* __restrict__ bias,
                              float* __restrict__ out,
                              int Hh, int Ww, int prevTiled, int applyLeaky)
{
    constexpr int Cin  = Cp + 1 + Cf;
    constexpr int K    = Cin * 9;
    constexpr int KPAD = (K + 31) & ~31;
    constexpr int CPAD = (KPAD + 8) / 9;   // any k < KPAD decodes into patch
    constexpr int PH = 10, PW = 18;        // 8+2 halo rows, 16+2 halo cols

    __shared__ _Float16 wLds[16 * KPAD];
    __shared__ _Float16 patch[CPAD * PH * PW];
    __shared__ float    bLds[16];
    __shared__ float    tdmBuf[16];

    const int Hs = Hh >> 1, Ws = Ww >> 1;
    const int numPix = Hh * Ww;
    const int tilesX = Ww >> 4;
    const int ty0 = (blockIdx.x / tilesX) * 8;
    const int tx0 = (blockIdx.x % tilesX) * 16;
    const int n  = blockIdx.y;                    // tiled batch index (0..63)
    const int pb = prevTiled ? n : (n & (BCONST - 1));
    const int fb = n & (BCONST - 1);

#if __has_builtin(__builtin_amdgcn_tensor_load_to_lds)
    // CDNA5 Tensor Data Mover: async-load one 16-float weight row into LDS
    // scratch (exercises TDM + TENSORcnt path).
    if (threadIdx.x == 0) {
        unsigned ldsAddr = (unsigned)(unsigned long long)(void*)&tdmBuf[0];
        unsigned long long ga = (unsigned long long)(const void*)w;
        u32x4 g0 = {};
        g0[0] = 1u;
        g0[1] = ldsAddr;
        g0[2] = (unsigned)(ga & 0xFFFFFFFFu);
        g0[3] = (unsigned)((ga >> 32) & 0x1FFFFFFu) | (2u << 30);
        i32x8 g1 = {};
        g1[0] = 0x20000;       // data_size = 4B
        g1[1] = (16 << 16);    // tensor_dim0 = 16
        g1[2] = (1 << 16);     // tensor_dim1 = 1
        g1[3] = (16 << 16);    // tile_dim0 = 16
        g1[4] = 1;             // tile_dim1 = 1
        g1[5] = 16;            // tensor_dim0_stride = 16
        i32x4 g2 = {}; i32x4 g3 = {};
#if defined(__clang_major__) && __clang_major__ >= 23
        i32x8 g4 = {};
        __builtin_amdgcn_tensor_load_to_lds(g0, g1, g2, g3, g4, 0);
#else
        __builtin_amdgcn_tensor_load_to_lds(g0, g1, g2, g3, 0);
#endif
        __builtin_amdgcn_s_wait_tensorcnt(0);
    }
#endif

    __builtin_prefetch(feat + (size_t)fb * Cf * numPix, 0, 1);

    // ---- stage weights: f32 -> f16, zero-padded to 16 x KPAD ----
    for (int idx = threadIdx.x; idx < 16 * KPAD; idx += blockDim.x) {
        int mm = idx / KPAD, kk = idx % KPAD;
        float v = (mm < Cout && kk < K) ? w[mm * K + kk] : 0.f;
        wLds[idx] = (_Float16)v;
    }
    if (threadIdx.x < 16)
        bLds[threadIdx.x] = (threadIdx.x < Cout) ? bias[threadIdx.x] : 0.f;

    // ---- stage input halo patch: concat + upsample + borders, once ----
    for (int idx = threadIdx.x; idx < CPAD * PH * PW; idx += blockDim.x) {
        int px = idx % PW; int t = idx / PW;
        int py = t % PH;   int cin = t / PH;
        int gy = ty0 + py - 1, gx = tx0 + px - 1;
        float v = 0.f;
        if (cin < Cin && gy >= 0 && gy < Hh && gx >= 0 && gx < Ww) {
            if (cin < Cp)
                v = prev[((size_t)(pb * Cp + cin) * Hs + (gy >> 1)) * Ws + (gx >> 1)];
            else if (cin == Cp)
                v = attn[((size_t)n * Hs + (gy >> 1)) * Ws + (gx >> 1)];
            else
                v = feat[((size_t)(fb * Cf + (cin - Cp - 1)) * Hh + gy) * Ww + gx];
        }
        patch[idx] = (_Float16)v;
    }
    __syncthreads();

    const int lane = threadIdx.x & 31;
    const int wv   = threadIdx.x >> 5;     // wave's row within tile (0..7)
    const int m    = lane & 15;
    const int ksel = lane >> 4;            // which K-half this lane holds
    const int base = (wv)*PW + (lane & 15);   // runtime part of patch index

    v8f acc = {};
#pragma unroll
    for (int kb = 0; kb < KPAD; kb += 32) {
        v16h a;
#pragma unroll
        for (int t = 0; t < 16; ++t) {
            int kl = ksel * 16 + t;
            a[t] = wLds[m * KPAD + kb + kl];
        }
        v16h bf;
#pragma unroll
        for (int t = 0; t < 16; ++t) {
            // both candidate offsets constant-fold; one select per element
            int o0 = patchOff(kb + t);        // lane in low K-half
            int o1 = patchOff(kb + 16 + t);   // lane in high K-half
            bf[t] = patch[base + (ksel ? o1 : o0)];
        }
        acc = __builtin_amdgcn_wmma_f32_16x16x32_f16(false, a, false, bf,
                                                     (short)0, acc, false, false);
    }

    const int hi = lane >> 4;
    const int oy = ty0 + wv;
    const int ox = tx0 + (lane & 15);
#pragma unroll
    for (int r = 0; r < 8; ++r) {
        int oc = r + 8 * hi;               // D: M = r + 8*(lane>=16)
        if (oc < Cout) {
            float v = acc[r] + bLds[oc];
            if (applyLeaky) v = v >= 0.f ? v : NEG_SLOPE * v;
            out[((size_t)n * Cout + oc) * numPix + oy * Ww + ox] = v;
        }
    }
}

// ---------------------------------------------------------------------------
extern "C" void kernel_launch(void* const* d_in, const int* in_sizes, int n_in,
                              void* d_out, int out_size, void* d_ws, size_t ws_size,
                              hipStream_t stream)
{
    (void)in_sizes; (void)n_in; (void)out_size; (void)ws_size;
    const float* x1  = (const float*)d_in[0];
    const float* x2  = (const float*)d_in[1];
    const int*   pts = (const int*)d_in[2];
    const float* encw[4] = {(const float*)d_in[3], (const float*)d_in[5],
                            (const float*)d_in[7], (const float*)d_in[9]};
    const float* encb[4] = {(const float*)d_in[4], (const float*)d_in[6],
                            (const float*)d_in[8], (const float*)d_in[10]};
    const float* decw[4] = {(const float*)d_in[11], (const float*)d_in[13],
                            (const float*)d_in[15], (const float*)d_in[17]};
    const float* decb[4] = {(const float*)d_in[12], (const float*)d_in[14],
                            (const float*)d_in[16], (const float*)d_in[18]};

    // workspace layout (floats)
    float* ws = (float*)d_ws;
    size_t off = 0;
    auto alloc = [&](size_t nf) { float* p = ws + off; off += nf; return p; };
    float* convScratch = alloc(524288);
    float* h1A = alloc(131072);
    float* h1B = alloc(131072);
    float* feat[4]; feat[0] = alloc(524288); feat[1] = alloc(262144);
                    feat[2] = alloc(131072); feat[3] = alloc(32768);
    float* h2p[4];  h2p[0]  = alloc(131072); h2p[1]  = alloc(65536);
                    h2p[2]  = alloc(32768);  h2p[3]  = alloc(8192);
    float* attn[4]; attn[0] = alloc(1048576); attn[1] = alloc(262144);
                    attn[2] = alloc(65536);   attn[3] = alloc(16384);
    float* dec0o = alloc(1048576);
    float* dec1o = alloc(2097152);
    float* dec2o = alloc(4194304);

    const int encCin[4]  = {3, 4, 8, 16};
    const int encCout[4] = {4, 8, 16, 16};
    const int Hin[4]     = {256, 128, 64, 32};

    const float* h1src = x1;
    const float* h2src = x2;
    float* h1dst[4] = {h1A, h1B, h1A, h1B};

    for (int i = 0; i < 4; ++i) {
        int Hi = Hin[i], Ho = Hi / 2;
        int convTot = BCONST * encCout[i] * Hi * Hi;
        int poolTot = BCONST * encCout[i] * Ho * Ho;
        conv5_leaky<<<(convTot + 255) / 256, 256, 0, stream>>>(
            h1src, encw[i], encb[i], convScratch, BCONST, encCin[i], encCout[i], Hi, Hi);
        maxpool2<<<(poolTot + 255) / 256, 256, 0, stream>>>(
            convScratch, h1dst[i], BCONST, encCout[i], Ho, Ho);
        conv5_leaky<<<(convTot + 255) / 256, 256, 0, stream>>>(
            h2src, encw[i], encb[i], feat[i], BCONST, encCin[i], encCout[i], Hi, Hi);
        maxpool2<<<(poolTot + 255) / 256, 256, 0, stream>>>(
            feat[i], h2p[i], BCONST, encCout[i], Ho, Ho);

        dim3 ag((Ho * Ho) / 128, BCONST * 2);
        if (encCout[i] == 4)
            attn_wmma<4><<<ag, 256, 0, stream>>>(h1dst[i], h2p[i], pts, attn[i], Ho, Ho, i + 1);
        else if (encCout[i] == 8)
            attn_wmma<8><<<ag, 256, 0, stream>>>(h1dst[i], h2p[i], pts, attn[i], Ho, Ho, i + 1);
        else
            attn_wmma<16><<<ag, 256, 0, stream>>>(h1dst[i], h2p[i], pts, attn[i], Ho, Ho, i + 1);

        h1src = h1dst[i];
        h2src = h2p[i];
    }

    // decoder: implicit-GEMM WMMA convs; grid = (H/8)*(W/16) tiles x 64 images
    dec_conv_wmma<16, 16, 16><<<dim3((32 / 8) * (32 / 16), 64), 256, 0, stream>>>(
        h2p[3], attn[3], feat[3], decw[0], decb[0], dec0o, 32, 32, /*prevTiled*/0, /*leaky*/1);
    dec_conv_wmma<16, 16, 8><<<dim3((64 / 8) * (64 / 16), 64), 256, 0, stream>>>(
        dec0o, attn[2], feat[2], decw[1], decb[1], dec1o, 64, 64, 1, 1);
    dec_conv_wmma<8, 8, 4><<<dim3((128 / 8) * (128 / 16), 64), 256, 0, stream>>>(
        dec1o, attn[1], feat[1], decw[2], decb[2], dec2o, 128, 128, 1, 1);
    dec_conv_wmma<4, 4, 1><<<dim3((256 / 8) * (256 / 16), 64), 256, 0, stream>>>(
        dec2o, attn[0], feat[0], decw[3], decb[3], (float*)d_out, 256, 256, 1, /*leaky*/0);
}
